// ResidualAdvection_26826365731354
// MI455X (gfx1250) — compile-verified, matrix operand order checked by reading, not run on hardware
//
#include <hip/hip_runtime.h>
#include <hip/hip_bf16.h>
#include <math.h>

// ---------------------------------------------------------------------------
// ResidualAdvection fused pipeline for MI455X (gfx1250, wave32, WMMA)
//   B=16, H=W=512
//   k0: pack conv weights into WMMA B-fragment layout (bf16) in d_ws
//   k1: conv1(3->16, 3x3) + exact GELU  -> feat NHWC bf16 in d_ws (128MB)
//   k2: async-stage feat halo to LDS (global_load_async_to_lds_b128),
//       conv2(16->18, 3x3) via 10x v_wmma_f32_16x16x32_bf16 per 16-pixel
//       tile (A from LDS), LDS transpose, 9-tap deformable bilinear
//       sampling of pm25, weighted sum -> out (B,1,H,W) fp32
// ---------------------------------------------------------------------------

#define BATCH 16
#define HDIM  512
#define WDIM  512
#define HW    (HDIM * WDIM)

typedef __attribute__((ext_vector_type(16))) __bf16 v16bf;
typedef __attribute__((ext_vector_type(8)))  float  v8f;
typedef __attribute__((ext_vector_type(4)))  unsigned int u32x4;
// Exact parameter type of the async-LDS builtins (gcc vector of int):
typedef int gi32x4 __attribute__((vector_size(16)));

#define GLOBAL_AS __attribute__((address_space(1)))
#define LDS_AS    __attribute__((address_space(3)))

#if __has_builtin(__builtin_amdgcn_global_load_async_to_lds_b128) && \
    __has_builtin(__builtin_amdgcn_s_wait_asynccnt)
#define RA_HAS_ASYNC 1
#else
#define RA_HAS_ASYNC 0
#endif

union AFrag {
    v16bf  v;
    u32x4  q[2];
    __bf16 h[16];
};

// ws layout (bytes):
//   [0,     1024)  : conv1 B fragment pack  (32 lanes * 16 halfs)
//   [1024, 11264)  : conv2 B fragment packs (5 chunks * 2 tiles * 512 halfs)
//   [16384, +128MB): feat, NHWC bf16  [b][y][x][c=16]
#define WS_PACK1_OFF 0
#define WS_PACK2_OFF 1024
#define WS_FEAT_OFF  16384

// ---------------------------------------------------------------------------
// Kernel 0: pack weights into the 16-bit WMMA fragment layout.
//   lane l, half j: khalf = l/16
//   K_local = (j<8) ? 8*khalf + j : 16 + 8*khalf + (j-8)
//   column  = l%16  (A: row M, B: col N)
// ---------------------------------------------------------------------------
__global__ void ra_pack_weights(const float* __restrict__ w1,
                                const float* __restrict__ w2,
                                __bf16* __restrict__ ws_bf16) {
    __bf16* pack1 = ws_bf16 + (WS_PACK1_OFF / 2);
    __bf16* pack2 = ws_bf16 + (WS_PACK2_OFF / 2);
    const int t = threadIdx.x;

    // conv1: K = neighbor*3 + ic  (27 real, pad to 32), N = out channel (16)
    for (int idx = t; idx < 32 * 16; idx += blockDim.x) {
        int lane = idx >> 4, j = idx & 15;
        int hi = lane >> 4, n = lane & 15;
        int K = (j < 8) ? (8 * hi + j) : (16 + 8 * hi + (j - 8));
        float v = 0.0f;
        if (K < 27) {
            int nb = K / 3, ic = K % 3, ky = nb / 3, kx = nb % 3;
            v = w1[((n * 3 + ic) * 3 + ky) * 3 + kx];   // OIHW (16,3,3,3)
        }
        pack1[lane * 16 + j] = (__bf16)v;
    }

    // conv2: K = neighbor*16 + ic (144 real, pad to 160), N = out ch (18 pad 32)
    for (int idx = t; idx < 5 * 2 * 512; idx += blockDim.x) {
        int ct = idx >> 9, rem = idx & 511;            // ct = c*2 + tile
        int c = ct >> 1, tile = ct & 1;
        int lane = rem >> 4, j = rem & 15;
        int hi = lane >> 4, ncol = tile * 16 + (lane & 15);
        int Kl = (j < 8) ? (8 * hi + j) : (16 + 8 * hi + (j - 8));
        int K = 32 * c + Kl;
        float v = 0.0f;
        if (K < 144 && ncol < 18) {
            int nb = K >> 4, ic = K & 15, ky = nb / 3, kx = nb % 3;
            v = w2[((ncol * 16 + ic) * 3 + ky) * 3 + kx];  // OIHW (18,16,3,3)
        }
        pack2[idx] = (__bf16)v;
    }
}

// ---------------------------------------------------------------------------
// Kernel 1: conv1 + GELU -> feat (NHWC bf16). One wave = 16 pixels of a row.
// block = 128 threads (4 waves), grid = (W/64, H, B)
// ---------------------------------------------------------------------------
__global__ void ra_conv1_gelu(const float* __restrict__ wind,
                              const float* __restrict__ topo,
                              const float* __restrict__ b1,
                              const __bf16* __restrict__ ws_bf16,
                              __bf16* __restrict__ feat) {
    __shared__ __align__(32) __bf16 sfeat[4][16][16];

    const int lane = threadIdx.x & 31;
    const int wave = threadIdx.x >> 5;
    const int hi = lane >> 4;         // K-half selector
    const int m = lane & 15;          // pixel within tile / output channel
    const int x0 = (blockIdx.x * 4 + wave) * 16;
    const int y = blockIdx.y;
    const int b = blockIdx.z;

    const float* windb = wind + (size_t)b * 2 * HW;
    const float* topob = topo + (size_t)b * HW;

    // Build A fragment: 16 pixels x K32 (K = neighbor*3 + ic, 27 real)
    AFrag a;
#pragma unroll
    for (int j = 0; j < 16; ++j) {
        int K = (j < 8) ? (8 * hi + j) : (16 + 8 * hi + (j - 8));
        float v = 0.0f;
        if (K < 27) {
            int nb = K / 3, ic = K % 3;
            int yy = y + nb / 3 - 1;
            int xx = x0 + m + nb % 3 - 1;
            if ((unsigned)yy < (unsigned)HDIM && (unsigned)xx < (unsigned)WDIM) {
                const float* src = (ic < 2) ? (windb + (size_t)ic * HW) : topob;
                v = src[yy * WDIM + xx];
            }
        }
        a.h[j] = (__bf16)v;
    }

    // B fragment from packed scratch
    v16bf bfrag = *(const v16bf*)(ws_bf16 + WS_PACK1_OFF / 2 + lane * 16);

    v8f acc = {};
    acc = __builtin_amdgcn_wmma_f32_16x16x32_bf16(
        false, a.v, false, bfrag, (short)0, acc, false, false);

    // bias + exact GELU; lane's column N = m is the output channel
    const float bias = b1[m];
#pragma unroll
    for (int r = 0; r < 8; ++r) {
        float x = acc[r] + bias;
        float g = 0.5f * x * (1.0f + erff(x * 0.70710678118654752f));
        sfeat[wave][r + 8 * hi][m] = (__bf16)g;   // [pixel][channel]
    }
    __syncthreads();

    // lanes 0..15: write one pixel's 16 channels as 2x16B vector stores
    if (lane < 16) {
        u32x4 q0 = *(const u32x4*)&sfeat[wave][lane][0];
        u32x4 q1 = *(const u32x4*)&sfeat[wave][lane][8];
        __bf16* dst = feat + (((size_t)b * HDIM + y) * WDIM + (x0 + lane)) * 16;
        *(u32x4*)(dst + 0) = q0;
        *(u32x4*)(dst + 8) = q1;
    }
}

// ---------------------------------------------------------------------------
// Kernel 2: async-stage the 3-row x 66-col feat halo to LDS, conv2 via WMMA
// (A fragments from LDS), then deformable 9-tap bilinear sampling.
// block = 128 threads (4 waves), grid = (W/64, H, B)
// ---------------------------------------------------------------------------
__device__ __forceinline__ float ra_fetch(const float* __restrict__ img,
                                          int y, int x) {
    if ((unsigned)y < (unsigned)HDIM && (unsigned)x < (unsigned)WDIM)
        return img[y * WDIM + x];
    return 0.0f;
}

__device__ __forceinline__ float ra_bilinear(const float* __restrict__ img,
                                             float py, float px) {
    float y0f = floorf(py), x0f = floorf(px);
    float wy = py - y0f, wx = px - x0f;
    int yi = (int)y0f, xi = (int)x0f;
    float v00 = ra_fetch(img, yi,     xi);
    float v01 = ra_fetch(img, yi,     xi + 1);
    float v10 = ra_fetch(img, yi + 1, xi);
    float v11 = ra_fetch(img, yi + 1, xi + 1);
    return v00 * (1.0f - wy) * (1.0f - wx) + v01 * (1.0f - wy) * wx
         + v10 * wy * (1.0f - wx)          + v11 * wy * wx;
}

__global__ void ra_conv2_sample(const float* __restrict__ pm25,
                                const float* __restrict__ b2,
                                const float* __restrict__ wk,
                                const __bf16* __restrict__ ws_bf16,
                                const __bf16* __restrict__ feat,
                                float* __restrict__ out) {
    // feat halo: rows y-1..y+1, cols xblk-1..xblk+64  (3 x 66 slots of 16ch)
    __shared__ __align__(16) __bf16 shfeat[3][68][16];
    __shared__ float soff[4][16][20];   // [wave][pixel][channel], padded

    const int tid  = threadIdx.x;
    const int lane = tid & 31;
    const int wave = tid >> 5;
    const int hi = lane >> 4;
    const int m = lane & 15;
    const int xblk = blockIdx.x * 64;
    const int x0 = xblk + wave * 16;
    const int y = blockIdx.y;
    const int b = blockIdx.z;

    const __bf16* featb = feat + (size_t)b * HW * 16;
    const float* pm25b = pm25 + (size_t)b * HW;

    // Warm pm25 rows for the sampling phase while staging + WMMAs run.
    {
        int py = min(max(y - 1, 0), HDIM - 1);
        __builtin_prefetch(pm25b + py * WDIM + x0, 0, 1);
        __builtin_prefetch(pm25b + y * WDIM + x0, 0, 1);
    }

    // ---- Stage halo: 3*66 pixel-slots, 32B each, cooperative across block.
    for (int slot = tid; slot < 3 * 66; slot += 128) {
        int r = slot / 66, cx = slot % 66;
        int gy = y - 1 + r;
        int gx = xblk - 1 + cx;
        if ((unsigned)gy < (unsigned)HDIM && (unsigned)gx < (unsigned)WDIM) {
            const __bf16* src = featb + (size_t)(gy * WDIM + gx) * 16;
#if RA_HAS_ASYNC
            __builtin_amdgcn_global_load_async_to_lds_b128(
                (GLOBAL_AS gi32x4*)(void*)src,
                (LDS_AS gi32x4*)(void*)&shfeat[r][cx][0], 0, 0);
            __builtin_amdgcn_global_load_async_to_lds_b128(
                (GLOBAL_AS gi32x4*)(void*)src,
                (LDS_AS gi32x4*)(void*)&shfeat[r][cx][0], 16, 0);
#else
            u32x4 lo = *(const u32x4*)(src + 0);
            u32x4 hiq = *(const u32x4*)(src + 8);
            *(u32x4*)&shfeat[r][cx][0] = lo;
            *(u32x4*)&shfeat[r][cx][8] = hiq;
#endif
        } else {
            *(u32x4*)&shfeat[r][cx][0] = (u32x4)0u;
            *(u32x4*)&shfeat[r][cx][8] = (u32x4)0u;
        }
    }

    const __bf16* pack2 = ws_bf16 + WS_PACK2_OFF / 2;

    // B fragments (broadcast, L2-resident) can issue while async fill runs.
    v16bf bt0[5], bt1[5];
#pragma unroll
    for (int c = 0; c < 5; ++c) {
        bt0[c] = *(const v16bf*)(pack2 + (size_t)(c * 2 + 0) * 512 + lane * 16);
        bt1[c] = *(const v16bf*)(pack2 + (size_t)(c * 2 + 1) * 512 + lane * 16);
    }

#if RA_HAS_ASYNC
    __builtin_amdgcn_s_wait_asynccnt(0);
#endif
    __syncthreads();

    v8f acc0 = {};   // output channels 0..15
    v8f acc1 = {};   // output channels 16..17 (rest zero-padded)

#pragma unroll
    for (int c = 0; c < 5; ++c) {
        // A chunk: K_local j<8 -> neighbor 2c, ch 8*hi..+7 ; j>=8 -> 2c+1
        AFrag a;
#pragma unroll
        for (int half = 0; half < 2; ++half) {
            int nb = 2 * c + half;
            u32x4 q = (u32x4)0u;
            if (nb < 9) {
                int r = nb / 3;
                int cx = wave * 16 + m + nb % 3;   // halo col for this pixel
                q = *(const u32x4*)&shfeat[r][cx][8 * hi];
            }
            a.q[half] = q;
        }
        acc0 = __builtin_amdgcn_wmma_f32_16x16x32_bf16(
            false, a.v, false, bt0[c], (short)0, acc0, false, false);
        acc1 = __builtin_amdgcn_wmma_f32_16x16x32_bf16(
            false, a.v, false, bt1[c], (short)0, acc1, false, false);
    }

    // bias + transpose through LDS: soff[pixel][channel]
    const float bias0 = b2[m];
    const float bias1 = (m < 2) ? b2[16 + m] : 0.0f;
#pragma unroll
    for (int r = 0; r < 8; ++r) {
        int px = r + 8 * hi;               // pixel index in tile
        soff[wave][px][m] = acc0[r] + bias0;
        if (m < 2) soff[wave][px][16 + m] = acc1[r] + bias1;
    }
    __syncthreads();

    // Sampling: lane handles pixel m; half-waves split taps 5/4.
    const int gx = x0 + m;
    float partial = 0.0f;
    for (int k = hi; k < 9; k += 2) {
        float dy = soff[wave][m][2 * k];
        float dx = soff[wave][m][2 * k + 1];
        float py = (float)y + (float)(k / 3 - 1) + dy;
        float px = (float)gx + (float)(k % 3 - 1) + dx;
        partial += wk[k] * ra_bilinear(pm25b, py, px);
    }
    partial += __shfl_xor(partial, 16, 32);
    if (lane < 16) {
        out[((size_t)b * HDIM + y) * WDIM + gx] = partial;
    }
}

// ---------------------------------------------------------------------------
// Host-side launch (graph-capture safe: all work on `stream`, no allocs)
// Requires ws_size >= WS_FEAT_OFF + B*H*W*16*2 bytes (~128 MB + 16 KB).
// ---------------------------------------------------------------------------
extern "C" void kernel_launch(void* const* d_in, const int* in_sizes, int n_in,
                              void* d_out, int out_size, void* d_ws, size_t ws_size,
                              hipStream_t stream) {
    (void)in_sizes; (void)n_in; (void)out_size; (void)ws_size;

    const float* pm25  = (const float*)d_in[0];
    const float* wind  = (const float*)d_in[1];
    const float* topo  = (const float*)d_in[2];
    const float* w1    = (const float*)d_in[3];
    const float* b1    = (const float*)d_in[4];
    const float* w2    = (const float*)d_in[5];
    const float* b2    = (const float*)d_in[6];
    const float* wk    = (const float*)d_in[7];   // (1,1,3,3) -> 9 taps
    float* out         = (float*)d_out;

    __bf16* ws_bf16 = (__bf16*)d_ws;
    __bf16* feat    = (__bf16*)((char*)d_ws + WS_FEAT_OFF);

    ra_pack_weights<<<1, 128, 0, stream>>>(w1, w2, ws_bf16);

    dim3 grid(WDIM / 64, HDIM, BATCH);
    dim3 block(128);
    ra_conv1_gelu<<<grid, block, 0, stream>>>(wind, topo, b1, ws_bf16, feat);
    ra_conv2_sample<<<grid, block, 0, stream>>>(pm25, b2, wk, ws_bf16, feat, out);
}
